// CA_SA_PA_global_local_40415642255801
// MI455X (gfx1250) — matrix operand order repaired and checked by reading, not compiled, not run on hardware
//
#include <hip/hip_runtime.h>
#include <stdint.h>

// ---------------- CDNA5 WMMA types ----------------
typedef __attribute__((ext_vector_type(16))) __bf16 v16bf;
typedef __attribute__((ext_vector_type(2)))  __bf16 v2bf;
typedef __attribute__((ext_vector_type(8)))  float  v8f;
typedef __attribute__((ext_vector_type(2)))  float  v2f;

struct Frag32 { uint4 a, b; };   // 32 bytes == v16bf

// f32 -> bf16 through the hardware convert (v_cvt_pk_bf16_f32).
__device__ __forceinline__ unsigned short f2bf(float f) {
  __bf16 h = (__bf16)f;
  return __builtin_bit_cast(unsigned short, h);
}
// packed pair conversion: one v_cvt_pk_bf16_f32, no mov/shift packing
__device__ __forceinline__ unsigned int pk2(float lo, float hi) {
  v2f f = {lo, hi};
  v2bf h = __builtin_convertvector(f, v2bf);
  return __builtin_bit_cast(unsigned int, h);
}
// 4 contiguous bf16 (8B) -> 4 floats; high halves expand for free (mask).
__device__ __forceinline__ void ld4bf(const unsigned short* p,
                                      float& a, float& b, float& c, float& d) {
  uint2 u = *(const uint2*)p;
  a = __uint_as_float(u.x << 16);
  b = __uint_as_float(u.x & 0xFFFF0000u);
  c = __uint_as_float(u.y << 16);
  d = __uint_as_float(u.y & 0xFFFF0000u);
}
// two separate 16B chunks -> one v16bf fragment
__device__ __forceinline__ v16bf load_frag2(const unsigned short* p0,
                                            const unsigned short* p1) {
  Frag32 f;
  f.a = *(const uint4*)p0;
  f.b = *(const uint4*)p1;
  return __builtin_bit_cast(v16bf, f);
}

#define QST 72   // transposed qkv row stride (64 positions + 8 pad), ushorts
                 // 72*2 = 144 bytes -> 16B-aligned rows for b128 C-stores

// One N-tile column of the QKV GEMM: D[64 x 16] += A[64 x K] * B[K x 16].
// A layout (ISA 7.12.2, 16-bit A 16x32): lane = (l15 row, lhb k-half),
//   chunk1 = K[kt*32 + lhb*8 .. +8), chunk2 = K[kt*32 + 16 + lhb*8 .. +8).
// B layout: lane holds column n = l15, same K chunks as A (lhb*8, 16 + lhb*8)
//   == row-major weight row W[n][k] -> load straight from global bf16 weights.
// C layout: VGPR r -> row (r + 8*lhb), col l15; the 8 rows are contiguous
//   positions in the transposed qkv layout -> one packed ds_store_b128.
template<int K>
__device__ __forceinline__ void gemm_col(const unsigned short* __restrict__ Wb,
                                         int nt,
                                         const unsigned short* __restrict__ A,
                                         int ast,
                                         unsigned short* __restrict__ QT,
                                         int l15, int lhb) {
  constexpr int KT = K / 32;
  v16bf bfr[KT];
  const unsigned short* wrow = Wb + (size_t)(nt * 16 + l15) * K + lhb * 8;
#pragma unroll
  for (int kt = 0; kt < KT; ++kt)
    bfr[kt] = load_frag2(wrow + kt * 32, wrow + kt * 32 + 16);
#pragma unroll
  for (int mt = 0; mt < 4; ++mt) {
    v8f acc = {0.f, 0.f, 0.f, 0.f, 0.f, 0.f, 0.f, 0.f};
    const unsigned short* arow = A + (mt * 16 + l15) * ast + lhb * 8;
#pragma unroll
    for (int kt = 0; kt < KT; ++kt) {
      v16bf afr = load_frag2(arow + kt * 32, arow + kt * 32 + 16);
      acc = __builtin_amdgcn_wmma_f32_16x16x32_bf16(
          false, afr, false, bfr[kt], (short)0, acc, false, false);
    }
    const int o = nt * 16 + l15;
    uint4 st;
    st.x = pk2(acc[0], acc[1]);
    st.y = pk2(acc[2], acc[3]);
    st.z = pk2(acc[4], acc[5]);
    st.w = pk2(acc[6], acc[7]);
    *(uint4*)(QT + o * QST + mt * 16 + lhb * 8) = st;
  }
}

// Channel attention for one (window, head): rows c0..c0+15 of a CHxCH score
// matrix with inner dim 4 (the 2x2 window taps). qkv stored transposed
// [o][p] so each q/k/v row is one contiguous 8-byte LDS load.
template<int CH>
__device__ __forceinline__ void attn_rows(const unsigned short* __restrict__ qT,
                                          unsigned short* __restrict__ oT,
                                          int win, int head, int c0,
                                          float scale) {
  const int Cch = CH * 4;     // channels per tensor (C)
  const int pb = win * 4;
  for (int c = c0; c < c0 + 16; ++c) {
    float q0, q1, q2, q3;
    ld4bf(qT + (c * 4 + head) * QST + pb, q0, q1, q2, q3);
    q0 *= scale; q1 *= scale; q2 *= scale; q3 *= scale;
    float s[CH];
    float mx = -3.4e38f;
#pragma unroll
    for (int d = 0; d < CH; ++d) {
      float k0, k1, k2, k3;
      ld4bf(qT + (Cch + d * 4 + head) * QST + pb, k0, k1, k2, k3);
      float t = q0 * k0 + q1 * k1 + q2 * k2 + q3 * k3;
      s[d] = t;
      mx = fmaxf(mx, t);
    }
    float sum = 0.f;
#pragma unroll
    for (int d = 0; d < CH; ++d) {
      float e = __expf(s[d] - mx);
      s[d] = e;
      sum += e;
    }
    float o0 = 0.f, o1 = 0.f, o2 = 0.f, o3 = 0.f;
#pragma unroll
    for (int d = 0; d < CH; ++d) {
      float v0, v1, v2, v3;
      ld4bf(qT + (2 * Cch + d * 4 + head) * QST + pb, v0, v1, v2, v3);
      const float pd = s[d];
      o0 += pd * v0; o1 += pd * v1; o2 += pd * v2; o3 += pd * v3;
    }
    const float inv = 1.0f / sum;
    uint2 st;
    st.x = pk2(o0 * inv, o1 * inv);
    st.y = pk2(o2 * inv, o3 * inv);
    *(uint2*)(oT + (c * 4 + head) * QST + pb) = st;
  }
}

// Prologue: convert f32 weights -> bf16 once (L2-resident afterwards).
__global__ void cvt_weights_kernel(const float* __restrict__ wlh,
                                   const float* __restrict__ wm,
                                   unsigned short* __restrict__ ws) {
  int i = blockIdx.x * 256 + threadIdx.x;
  if (i < 12288) ws[i] = f2bf(wlh[i]);            // 192x64
  if (i < 49152) ws[12288 + i] = f2bf(wm[i]);     // 384x128
}

// Fused: DWT -> 3x QKV WMMA GEMM -> windowed channel attention -> IDWT.
// Grid (4, 64, 8); block 256 (8 wave32). Each WG: 16 windows (4x64 pixels).
__global__ __launch_bounds__(256)
void CA_SA_PA_global_local_40415642255801_kernel(
    const float* __restrict__ x,
    const unsigned short* __restrict__ wlh,   // bf16 [192][64]
    const unsigned short* __restrict__ wm,    // bf16 [384][128]
    float* __restrict__ out) {
  __shared__ __align__(16) unsigned char smem[147456];
  unsigned short* qlowT  = (unsigned short*)(smem);           // 192 x 72
  unsigned short* qmidT  = (unsigned short*)(smem + 27648);   // 384 x 72
  unsigned short* qhighT = (unsigned short*)(smem + 82944);   // 192 x 72
  unsigned short* A_ll   = (unsigned short*)(smem + 110592);  // 64 x 72
  unsigned short* A_mid  = (unsigned short*)(smem + 119808);  // 64 x 136
  unsigned short* A_hh   = (unsigned short*)(smem + 138240);  // 64 x 72
  unsigned short* o_lowT  = A_ll;    // reuse after GEMM: 64 x 72
  unsigned short* o_midT  = A_mid;   // 128 x 72 (slot is 18432 B)
  unsigned short* o_highT = A_hh;    // 64 x 72

  const int tx = blockIdx.x, wy = blockIdx.y, b = blockIdx.z;
  const int tid = threadIdx.x;

  // ---- Stage 1: load pixels + Haar DWT -> bf16 A matrices (window-major p)
  {
    const int ch = tid >> 2, dy = (tid >> 1) & 1, hf = tid & 1;
    const int pr0 = wy * 4 + dy * 2;
    const int pc0 = tx * 64 + hf * 32;
    const float* src = x + (((size_t)(b * 64 + ch)) * 256 + pr0) * 256 + pc0;
    float r0[32], r1[32];
#pragma unroll
    for (int q = 0; q < 8; ++q) {
      float4 t0 = *(const float4*)(src + q * 4);
      float4 t1 = *(const float4*)(src + 256 + q * 4);
      r0[q*4+0]=t0.x; r0[q*4+1]=t0.y; r0[q*4+2]=t0.z; r0[q*4+3]=t0.w;
      r1[q*4+0]=t1.x; r1[q*4+1]=t1.y; r1[q*4+2]=t1.z; r1[q*4+3]=t1.w;
    }
#pragma unroll
    for (int j = 0; j < 16; ++j) {
      const float a = r0[2*j], bb = r0[2*j+1], c = r1[2*j], d = r1[2*j+1];
      const float ll = (a+bb+c+d)*0.5f, lh = (a-bb+c-d)*0.5f;
      const float hl = (a+bb-c-d)*0.5f, hh = (a-bb-c+d)*0.5f;
      const int qc = hf * 16 + j;
      const int p = (qc >> 1) * 4 + dy * 2 + (qc & 1);   // win*4 + sy*2 + sx
      A_ll [p * 72 + ch]       = f2bf(ll);
      A_mid[p * 136 + ch]      = f2bf(lh);
      A_mid[p * 136 + 64 + ch] = f2bf(hl);
      A_hh [p * 72 + ch]       = f2bf(hh);
    }
  }
  __syncthreads();

  // ---- Stage 2: QKV projections as WMMA bf16 GEMMs (48 N-columns / 8 waves)
  {
    const int lane = tid & 31, wv = tid >> 5;
    const int l15 = lane & 15, lhb = lane >> 4;
    for (int i = 0; i < 6; ++i) {
      const int col = wv * 6 + i;
      if (col < 12)      gemm_col<64>(wlh, col,      A_ll,  72,  qlowT,  l15, lhb);
      else if (col < 36) gemm_col<128>(wm, col - 12, A_mid, 136, qmidT,  l15, lhb);
      else               gemm_col<64>(wlh, col - 36, A_hh,  72,  qhighT, l15, lhb);
    }
  }
  __syncthreads();

  // ---- Stage 3: per-(window, head) channel attention (ch=16 low/high, 32 mid)
  {
    if (tid < 64) {
      attn_rows<16>(qlowT, o_lowT, tid >> 2, tid & 3, 0, 0.25f);
    } else if (tid < 128) {
      const int t = tid - 64;
      attn_rows<16>(qhighT, o_highT, t >> 2, t & 3, 0, 0.25f);
    } else {
      const int t = tid - 128;  // 16 win x 4 heads x 2 half-row-blocks
      attn_rows<32>(qmidT, o_midT, t >> 3, (t >> 1) & 3, (t & 1) * 16,
                    0.17677669529663687f);
    }
  }
  __syncthreads();

  // ---- Stage 4: inverse Haar + coalesced float4 stores
  {
    const int ch = tid >> 2, dy = (tid >> 1) & 1, hf = tid & 1;
    float r0[32], r1[32];
#pragma unroll
    for (int t = 0; t < 8; ++t) {
      const int win = hf * 8 + t;
      const int pbase = win * 4 + dy * 2;        // positions p, p+1 (dx=0,1)
      const unsigned int ull = *(const unsigned int*)(o_lowT  + ch * QST + pbase);
      const unsigned int ulh = *(const unsigned int*)(o_midT  + ch * QST + pbase);
      const unsigned int uhl = *(const unsigned int*)(o_midT  + (64 + ch) * QST + pbase);
      const unsigned int uhh = *(const unsigned int*)(o_highT + ch * QST + pbase);
      // dx = 0 (low halves)
      {
        const float ll = __uint_as_float(ull << 16);
        const float lh = __uint_as_float(ulh << 16);
        const float hl = __uint_as_float(uhl << 16);
        const float hh = __uint_as_float(uhh << 16);
        r0[4*t+0] = (ll + lh + hl + hh) * 0.5f;
        r0[4*t+1] = (ll - lh + hl - hh) * 0.5f;
        r1[4*t+0] = (ll + lh - hl - hh) * 0.5f;
        r1[4*t+1] = (ll - lh - hl + hh) * 0.5f;
      }
      // dx = 1 (high halves)
      {
        const float ll = __uint_as_float(ull & 0xFFFF0000u);
        const float lh = __uint_as_float(ulh & 0xFFFF0000u);
        const float hl = __uint_as_float(uhl & 0xFFFF0000u);
        const float hh = __uint_as_float(uhh & 0xFFFF0000u);
        r0[4*t+2] = (ll + lh + hl + hh) * 0.5f;
        r0[4*t+3] = (ll - lh + hl - hh) * 0.5f;
        r1[4*t+2] = (ll + lh - hl - hh) * 0.5f;
        r1[4*t+3] = (ll - lh - hl + hh) * 0.5f;
      }
    }
    const int pr0 = wy * 4 + dy * 2;
    const int pc0 = tx * 64 + hf * 32;
    float* dst = out + (((size_t)(b * 64 + ch)) * 256 + pr0) * 256 + pc0;
#pragma unroll
    for (int q = 0; q < 8; ++q) {
      float4 s0, s1;
      s0.x = r0[q*4+0]; s0.y = r0[q*4+1]; s0.z = r0[q*4+2]; s0.w = r0[q*4+3];
      s1.x = r1[q*4+0]; s1.y = r1[q*4+1]; s1.z = r1[q*4+2]; s1.w = r1[q*4+3];
      *(float4*)(dst + q * 4) = s0;
      *(float4*)(dst + 256 + q * 4) = s1;
    }
  }
}

extern "C" void kernel_launch(void* const* d_in, const int* in_sizes, int n_in,
                              void* d_out, int out_size, void* d_ws, size_t ws_size,
                              hipStream_t stream) {
  const float* x    = (const float*)d_in[0];   // (8,64,256,256)
  const float* w_lh = (const float*)d_in[1];   // (192,64)
  const float* w_m  = (const float*)d_in[2];   // (384,128)
  unsigned short* wbf = (unsigned short*)d_ws; // [0,12288): w_lh bf16, then w_m

  cvt_weights_kernel<<<192, 256, 0, stream>>>(w_lh, w_m, wbf);

  dim3 grid(4, 64, 8);   // (window-col tile, window row, batch)
  CA_SA_PA_global_local_40415642255801_kernel<<<grid, 256, 0, stream>>>(
      x, wbf, wbf + 12288, (float*)d_out);
}
